// MMGNNLayer_13254269075914
// MI455X (gfx1250) — matrix operand order repaired and checked by reading, not compiled, not run on hardware
//
#include <hip/hip_runtime.h>
#include <stdint.h>

// ---------------- problem constants (from reference) ----------------
#define NN    4096
#define DD    256
#define HH    4
#define DHD   64
#define ALPHA 0.2f
#define NEGV  (-9.0e15f)
#define MW    128           // mask words per row = NN/32
#define HROW  72            // padded LDS row stride in halfs (64 data + 8 pad)

typedef __attribute__((ext_vector_type(16))) _Float16 v16h;
typedef __attribute__((ext_vector_type(8)))  float    v8f;
typedef __attribute__((ext_vector_type(4)))  unsigned v4u;
typedef __attribute__((ext_vector_type(8)))  int      v8i;
typedef __attribute__((ext_vector_type(4)))  int      v4i;

#if defined(__has_builtin)
#if __has_builtin(__builtin_amdgcn_tensor_load_to_lds)
#define HAS_TDM 1
#endif
#endif

// ---------------- kernel 0: zero the adjacency bitmask ----------------
__global__ void k_zero_mask(unsigned* __restrict__ mask) {
  int t = blockIdx.x * blockDim.x + threadIdx.x;     // exactly NN*MW threads
  mask[t] = 0u;
}

// ---------------- kernel 1: scatter mask bits (MLP is dead code: sigmoid>0) ----------------
__global__ void k_mask(const int* __restrict__ ei, const float* __restrict__ ev,
                       unsigned* __restrict__ mask, int E) {
  int e = blockIdx.x * blockDim.x + threadIdx.x;
  if (e >= E) return;
  if (ev[e] > 0.0f) {                                 // adj>0  <=>  edge_vals>0
    int r = ei[e];
    int c = ei[E + e];
    atomicOr(&mask[r * MW + (c >> 5)], 1u << (c & 31));
  }
}

// ---------------- kernel 2: fold attention vectors through W ----------------
__global__ void k_fold(const float* __restrict__ W, const float* __restrict__ a_src,
                       const float* __restrict__ a_dst, float* __restrict__ cvec) {
  int t = blockIdx.x * blockDim.x + threadIdx.x;      // 8*DD threads
  if (t >= 8 * DD) return;
  int v = t / DD, d = t % DD;
  int hd = v & 3;
  const float* a = (v < 4) ? (a_src + hd * DHD) : (a_dst + hd * DHD);
  const float* w = W + (size_t)hd * DD * DHD + (size_t)d * DHD;
  float s = 0.f;
#pragma unroll 8
  for (int k = 0; k < DHD; ++k) s += w[k] * a[k];
  cvec[v * DD + d] = s;
}

// ---------------- kernel 3: esrc/edst = x @ cvec ----------------
__global__ void k_eproj(const float* __restrict__ x, const float* __restrict__ cvec,
                        float* __restrict__ esrc, float* __restrict__ edst) {
  int t = blockIdx.x * blockDim.x + threadIdx.x;      // 8*NN threads
  int v = t >> 12;                                    // 0..7
  int n = t & (NN - 1);
  const float* xr = x + (size_t)n * DD;
  const float* cv = cvec + v * DD;
  float s = 0.f;
#pragma unroll 8
  for (int d = 0; d < DD; ++d) s += xr[d] * cv[d];
  if (v < 4) esrc[v * NN + n] = s;
  else       edst[(v - 4) * NN + n] = s;
}

// ---------------- kernel 4: hF16 = f16(x @ Wcat)  via WMMA ----------------
__global__ void __launch_bounds__(32)
k_hgemm(const float* __restrict__ x, const float* __restrict__ W,
        _Float16* __restrict__ hF16) {
  int tile = blockIdx.x;                 // (NN/16) * (256/16) = 4096 tiles
  int i0   = (tile >> 4) << 4;
  int c0   = (tile & 15) << 4;
  int lane = threadIdx.x;
  int row  = lane & 15;
  int hi   = lane >> 4;
  int n    = lane & 15;
  int hd   = c0 >> 6;
  int kk0  = c0 & 63;

  v8f acc = {};
  const float* xr = x + (size_t)(i0 + row) * DD;
  const float* wb = W + (size_t)hd * DD * DHD + (kk0 + n);

  for (int k0 = 0; k0 < DD; k0 += 32) {
    v16h a, b;
#pragma unroll
    for (int i = 0; i < 16; ++i) {
      int K = hi * 16 + i;                                         // contiguous k = 16*hi + i A-frag order
      a[i] = (_Float16)xr[k0 + K];
    }
#pragma unroll
    for (int i = 0; i < 16; ++i) {
      int Kb = hi * 16 + i;                                        // contiguous k = 16*hi + i B-frag order
      b[i] = (_Float16)wb[(size_t)(k0 + Kb) * DHD];
    }
    acc = __builtin_amdgcn_wmma_f32_16x16x32_f16(false, a, false, b,
                                                 (short)0, acc, false, false);
  }
#pragma unroll
  for (int r = 0; r < 8; ++r) {
    int M = hi * 8 + r;                                            // ISA C/D layout
    hF16[(size_t)(i0 + M) * (HH * DHD) + c0 + n] = (_Float16)acc[r];
  }
}

#if HAS_TDM
// Issue one TDM descriptor: 32 rows x 64 halfs tile, global row stride 256 halfs,
// LDS destination padded to HROW halfs/row via pad_interval=32 DWORDs, pad_amount=4 DWORDs.
__device__ __forceinline__ void tdm_load_chunk(const _Float16* gsrc, unsigned lds_off) {
  unsigned long long ga = (unsigned long long)(uintptr_t)gsrc;
  v4u g0;
  g0.x = 0x1u;                                        // count=1 (valid user D#)
  g0.y = lds_off;                                     // lds_addr (bytes)
  g0.z = (unsigned)(ga & 0xFFFFFFFFu);                // global_addr[31:0]
  g0.w = (unsigned)((ga >> 32) & 0x01FFFFFFu)         // global_addr[56:32]
       | (2u << 30);                                  // type = 2 ("image")
  const unsigned TD0 = 1u << 20, TD1 = 1u << 20;      // generous tensor dims (no OOB)
  v8i g1;
  g1[0] = (int)((1u << 16)                            // data_size = 1 -> 2 bytes
        | (1u << 20)                                  // pad_enable
        | (4u << 22)                                  // pad_interval: 32 DWORDs (128B row)
        | (3u << 25));                                // pad_amount: 4 DWORDs (8 halfs)
  g1[1] = (int)((TD0 & 0xFFFFu) << 16);               // tensor_dim0[15:0]
  g1[2] = (int)(((TD0 >> 16) & 0xFFFFu) | ((TD1 & 0xFFFFu) << 16));
  g1[3] = (int)(((TD1 >> 16) & 0xFFFFu) | (64u << 16)); // tile_dim0 = 64
  g1[4] = 32;                                         // tile_dim1 = 32, tile_dim2 = 0
  g1[5] = 256;                                        // tensor_dim0_stride = 256 elems
  g1[6] = 0;
  g1[7] = 0;
  v4i z4 = {0, 0, 0, 0};
  v8i z8 = {0, 0, 0, 0, 0, 0, 0, 0};
  __builtin_amdgcn_tensor_load_to_lds(g0, g1, z4, z4, z8, 0);
}
#endif

// ---------------- kernel 5: masked softmax + attn@h ----------------
__global__ void __launch_bounds__(32)
k_attn(const unsigned* __restrict__ mask, const float* __restrict__ esrc,
       const float* __restrict__ edst, const _Float16* __restrict__ hF16,
       float* __restrict__ out) {
  __shared__ float s_m[16], s_l[16], s_es[16], s_ed[32];
  __shared__ _Float16 s_h[2][32 * HROW];   // double-buffered h chunks

  int b    = blockIdx.x;                   // HH * (NN/16) = 1024
  int hd   = b >> 8;
  int i0   = (b & 255) << 4;
  int lane = threadIdx.x;
  int row  = lane & 15;
  int hi   = lane >> 4;
  int n    = lane & 15;

  const float* edh = edst + hd * NN;
  if (lane < 16) s_es[lane] = esrc[hd * NN + i0 + lane];
  __syncthreads();

  // ---- pass 1: per-row online (max, sumexp) over all j ----
  for (int r = 0; r < 16; ++r) {
    float es = s_es[r];
    const unsigned* mrow = mask + (size_t)(i0 + r) * MW;
    float m = -INFINITY, l = 0.f;
    for (int t = 0; t < MW; ++t) {
      unsigned w = mrow[t];
      float e = es + edh[t * 32 + lane];
      e = (e > 0.f) ? e : ALPHA * e;                  // leaky_relu
      e = ((w >> lane) & 1u) ? e : NEGV;              // mask fill
      float mn = fmaxf(m, e);
      l = l * __expf(m - mn) + __expf(e - mn);
      m = mn;
    }
#pragma unroll
    for (int off = 16; off; off >>= 1) {              // wave32 (m,l) combine
      float m2 = __shfl_xor(m, off, 32);
      float l2 = __shfl_xor(l, off, 32);
      float mn = fmaxf(m, m2);
      l = l * __expf(m - mn) + l2 * __expf(m2 - mn);
      m = mn;
    }
    if (lane == 0) { s_m[r] = m; s_l[r] = l; }
  }
  __syncthreads();

  // ---- pass 2: out_tile = sum_j exp(e-m) * h_j  (WMMA f32 accum) ----
  float m_r  = s_m[row];
  float es_r = s_es[row];
  v8f c[4] = {{}, {}, {}, {}};

  const _Float16* gsrc = hF16 + hd * DHD;             // + j*256 per row j

#if HAS_TDM
  unsigned ldsoff[2] = { (unsigned)(uintptr_t)(void*)&s_h[0][0],
                         (unsigned)(uintptr_t)(void*)&s_h[1][0] };
  tdm_load_chunk(gsrc, ldsoff[0]);                    // preload chunk 0
#endif

  for (int j0 = 0; j0 < NN; j0 += 32) {
    int buf = (j0 >> 5) & 1;
#if HAS_TDM
    if (j0 + 32 < NN) {                               // issue next, wait current
      tdm_load_chunk(gsrc + (size_t)(j0 + 32) * (HH * DHD), ldsoff[buf ^ 1]);
      __builtin_amdgcn_s_wait_tensorcnt(1);           // in-order: current chunk done
    } else {
      __builtin_amdgcn_s_wait_tensorcnt(0);
    }
#else
    { // manual staging fallback: each lane copies one h row (4x b128)
      const uint4* hv = (const uint4*)(gsrc + (size_t)(j0 + lane) * (HH * DHD));
      uint4* dsrow = (uint4*)&s_h[buf][lane * HROW];
      dsrow[0] = hv[0]; dsrow[1] = hv[1]; dsrow[2] = hv[2]; dsrow[3] = hv[3];
    }
#endif
    s_ed[lane] = edh[j0 + lane];
    __syncthreads();

    if (j0 + 32 < NN)                                  // global_prefetch_b8
      __builtin_prefetch(gsrc + (size_t)(j0 + 32 + lane) * (HH * DHD), 0, 1);

    // A fragment: probabilities-hat, A order is contiguous k = 16*hi + i
    unsigned w = mask[(size_t)(i0 + row) * MW + (j0 >> 5)];
    v16h a;
#pragma unroll
    for (int i = 0; i < 16; ++i) {
      int K = hi * 16 + i;
      float e = es_r + s_ed[K];
      e = (e > 0.f) ? e : ALPHA * e;
      e = ((w >> K) & 1u) ? e : NEGV;
      a[i] = (_Float16)__expf(e - m_r);                // masked -> exp(-9e15-m) == 0
    }

#pragma unroll
    for (int kt = 0; kt < 4; ++kt) {
      v16h bb;
#pragma unroll
      for (int i = 0; i < 16; ++i)
        bb[i] = s_h[buf][(hi * 16 + i) * HROW + kt * 16 + n]; // contiguous-k B gather (k = 16*hi + i)
      c[kt] = __builtin_amdgcn_wmma_f32_16x16x32_f16(false, a, false, bb,
                                                     (short)0, c[kt], false, false);
    }
    __syncthreads();
  }

  // epilogue: divide by row sum, write [N, H*DH]
#pragma unroll
  for (int kt = 0; kt < 4; ++kt) {
#pragma unroll
    for (int r = 0; r < 8; ++r) {
      int M = hi * 8 + r;
      out[(size_t)(i0 + M) * (HH * DHD) + hd * DHD + kt * 16 + n] = c[kt][r] / s_l[M];
    }
  }
}

// ---------------- host-side launch ----------------
extern "C" void kernel_launch(void* const* d_in, const int* in_sizes, int n_in,
                              void* d_out, int out_size, void* d_ws, size_t ws_size,
                              hipStream_t stream) {
  const float* x     = (const float*)d_in[0];
  const float* ev    = (const float*)d_in[1];
  // d_in[2..5] = W1,b1,W2,b2  -- dead code: only sign of sigmoid output matters, always > 0
  const float* W     = (const float*)d_in[6];
  const float* a_src = (const float*)d_in[7];
  const float* a_dst = (const float*)d_in[8];
  const int*   ei    = (const int*)d_in[9];
  const int    E     = in_sizes[9] / 2;

  char* ws = (char*)d_ws;
  size_t maskBytes = (size_t)NN * MW * 4;       // 2 MB
  size_t hBytes    = (size_t)NN * HH * DHD * 2; // 2 MB
  unsigned* mask   = (unsigned*)ws;
  _Float16* hF16   = (_Float16*)(ws + maskBytes);
  float*    esrc   = (float*)(ws + maskBytes + hBytes);
  float*    edst   = (float*)(ws + maskBytes + hBytes + (size_t)HH * NN * 4);
  float*    cvec   = (float*)(ws + maskBytes + hBytes + (size_t)2 * HH * NN * 4);
  float*    out    = (float*)d_out;

  k_zero_mask<<<(NN * MW) / 256, 256, 0, stream>>>(mask);
  k_mask<<<(E + 255) / 256, 256, 0, stream>>>(ei, ev, mask, E);
  k_fold<<<(8 * DD + 255) / 256, 256, 0, stream>>>(W, a_src, a_dst, cvec);
  k_eproj<<<(8 * NN) / 256, 256, 0, stream>>>(x, cvec, esrc, edst);
  k_hgemm<<<(NN / 16) * ((HH * DHD) / 16), 32, 0, stream>>>(x, W, hF16);
  k_attn<<<HH * (NN / 16), 32, 0, stream>>>(mask, esrc, edst, hF16, out);
}
